// DeformLayer_41721312313801
// MI455X (gfx1250) — matrix operand order repaired and checked by reading, not compiled, not run on hardware
//
#include <hip/hip_runtime.h>
#include <hip/hip_bf16.h>
#include <math.h>

typedef __attribute__((ext_vector_type(16))) _Float16 v16h;
typedef __attribute__((ext_vector_type(8)))  _Float16 v8h;
typedef __attribute__((ext_vector_type(8)))  float    v8f;

#define NB    4
#define HH    64
#define WW    64
#define CC    256
#define GG    8
#define PP    9
#define HWSZ  4096                 // HH*WW
#define MROWS 16384                // NB*HH*WW

// ---------------------------------------------------------------------------
// Weight prep: w (K x Nact, row-major f32) -> Bt (Npad x K, row-major f16),
// zero-padded rows for n >= Nact.  Npad is a multiple of 32.
// ---------------------------------------------------------------------------
__global__ void prep_w(const float* __restrict__ w, _Float16* __restrict__ bt,
                       int Nact, int Npad, int K) {
    int idx = blockIdx.x * blockDim.x + threadIdx.x;
    if (idx >= Npad * K) return;
    int n = idx / K;
    int k = idx - n * K;
    bt[idx] = (n < Nact) ? (_Float16)w[(size_t)k * Nact + n] : (_Float16)0.f;
}

// ---------------------------------------------------------------------------
// NCHW -> NHWC, fp32 + fp16 copies
// ---------------------------------------------------------------------------
__global__ void nchw_to_nhwc(const float* __restrict__ in,
                             float* __restrict__ xf, _Float16* __restrict__ xh) {
    int idx = blockIdx.x * blockDim.x + threadIdx.x;   // n*C*HW + c*HW + hw
    if (idx >= NB * CC * HWSZ) return;
    int hw = idx & (HWSZ - 1);
    int c  = (idx >> 12) & (CC - 1);
    int n  = idx >> 20;
    float v = in[idx];
    size_t o = ((size_t)(n * HWSZ + hw)) * CC + c;
    xf[o] = v;
    xh[o] = (_Float16)v;
}

// ---------------------------------------------------------------------------
// Fragment loaders for v_wmma_f32_16x16x32_f16 (wave32 layouts, ISA 7.12.2):
//   A 16x32: lane L<16 -> M=L, K={0..7,16..23}; L>=16 -> K={8..15,24..31}
//   B 32x16: lane L<16 -> N=L, K={0..7,16..23}; L>=16 -> K={8..15,24..31}
// ---------------------------------------------------------------------------
__device__ __forceinline__ v16h load_afrag(const _Float16* p) {
    v8h a0 = *(const v8h*)(p);
    v8h a1 = *(const v8h*)(p + 16);
    v16h av;
#pragma unroll
    for (int i = 0; i < 8; ++i) { av[i] = a0[i]; av[8 + i] = a1[i]; }
    return av;
}
__device__ __forceinline__ v16h load_bfrag(const _Float16* p) {
    v8h b0 = *(const v8h*)(p);
    v8h b1 = *(const v8h*)(p + 16);
    v16h bv;
#pragma unroll
    for (int i = 0; i < 8; ++i) { bv[i] = b0[i]; bv[8 + i] = b1[i]; }
    return bv;
}

// ---------------------------------------------------------------------------
// WMMA GEMM, 2x2 register blocking: each wave computes a 32x32 output
// macro-tile (4 accumulators), 8 b128 loads feed 4 WMMAs per k-step.
// C[M x Nact] = A[M x K](f16) * Bt[Npad x K]^T(f16) + bias.
// nchw!=0: scatter output to NCHW (C=256, HW=4096).
// ---------------------------------------------------------------------------
__global__ __launch_bounds__(256)
void gemm_wmma_f16(const _Float16* __restrict__ A, const _Float16* __restrict__ Bt,
                   const float* __restrict__ bias, float* __restrict__ C,
                   int Nact, int Nt2, int Kdim, int nchw) {
    const int wave = threadIdx.x >> 5;
    const int lane = threadIdx.x & 31;
    const int t    = blockIdx.x * 8 + wave;
    const int mt   = t / Nt2;
    const int nt   = t - mt * Nt2;
    const int m0   = mt << 5;          // 32-row macro tile
    const int n0   = nt << 5;          // 32-col macro tile
    const int hs   = lane >> 4;        // 0: low half, 1: high half
    const int lrow = lane & 15;

    const _Float16* Ap0 = A  + (size_t)(m0 + lrow) * Kdim + hs * 8;
    const _Float16* Ap1 = Ap0 + (size_t)16 * Kdim;
    const _Float16* Bp0 = Bt + (size_t)(n0 + lrow) * Kdim + hs * 8;
    const _Float16* Bp1 = Bp0 + (size_t)16 * Kdim;

    v8f acc00 = {}, acc01 = {}, acc10 = {}, acc11 = {};
    for (int k = 0; k < Kdim; k += 32) {
        __builtin_prefetch(Ap0 + k + 128, 0, 0);
        __builtin_prefetch(Ap1 + k + 128, 0, 0);
        __builtin_prefetch(Bp0 + k + 128, 0, 0);
        __builtin_prefetch(Bp1 + k + 128, 0, 0);
        v16h a0 = load_afrag(Ap0 + k);
        v16h a1 = load_afrag(Ap1 + k);
        v16h b0 = load_bfrag(Bp0 + k);
        v16h b1 = load_bfrag(Bp1 + k);
        acc00 = __builtin_amdgcn_wmma_f32_16x16x32_f16(false, a0, false, b0, (short)0, acc00, false, false);
        acc01 = __builtin_amdgcn_wmma_f32_16x16x32_f16(false, a0, false, b1, (short)0, acc01, false, false);
        acc10 = __builtin_amdgcn_wmma_f32_16x16x32_f16(false, a1, false, b0, (short)0, acc10, false, false);
        acc11 = __builtin_amdgcn_wmma_f32_16x16x32_f16(false, a1, false, b1, (short)0, acc11, false, false);
    }

    // Epilogue: D layout (VGPR r -> M = r + hs*8, N = lane&15)
#pragma unroll
    for (int tj = 0; tj < 2; ++tj) {
        const int col = n0 + tj * 16 + lrow;
        if (col >= Nact) continue;
        const float bb = bias[col];
#pragma unroll
        for (int ti = 0; ti < 2; ++ti) {
            const v8f acc = (tj == 0) ? (ti == 0 ? acc00 : acc10)
                                      : (ti == 0 ? acc01 : acc11);
#pragma unroll
            for (int r = 0; r < 8; ++r) {
                const int row = m0 + ti * 16 + r + hs * 8;
                const float v = acc[r] + bb;
                if (!nchw) {
                    C[(size_t)row * Nact + col] = v;
                } else {
                    const int n  = row >> 12;
                    const int hw = row & (HWSZ - 1);
                    C[((size_t)(n * CC + col)) * HWSZ + hw] = v;
                }
            }
        }
    }
}

// ---------------------------------------------------------------------------
// Fused depthwise 3x3 conv + LayerNorm(C) + exact GELU.
// One block (256 threads) per pixel; thread = channel. LDS reduction.
// ---------------------------------------------------------------------------
__global__ __launch_bounds__(256)
void conv_ln_gelu(const float* __restrict__ x, const float* __restrict__ dwk,
                  const float* __restrict__ dwb, const float* __restrict__ lng,
                  const float* __restrict__ lnb,
                  float* __restrict__ x1, _Float16* __restrict__ x1h) {
    const int pix = blockIdx.x;
    const int c   = threadIdx.x;
    const int w   = pix & 63;
    const int h   = (pix >> 6) & 63;
    const int n   = pix >> 12;

    float s = dwb[c];
#pragma unroll
    for (int ky = 0; ky < 3; ++ky) {
        int yy = h + ky - 1;
        if ((unsigned)yy >= (unsigned)HH) continue;
#pragma unroll
        for (int kx = 0; kx < 3; ++kx) {
            int xx = w + kx - 1;
            if ((unsigned)xx >= (unsigned)WW) continue;
            s += x[((size_t)((n * HH + yy) * WW + xx)) * CC + c]
               * dwk[(ky * 3 + kx) * CC + c];
        }
    }

    __shared__ float rs[8], rq[8];
    __shared__ float smu, sivar;
    float v = s, q = s * s;
#pragma unroll
    for (int o = 16; o > 0; o >>= 1) {
        v += __shfl_down(v, o, 32);
        q += __shfl_down(q, o, 32);
    }
    if ((threadIdx.x & 31) == 0) { rs[threadIdx.x >> 5] = v; rq[threadIdx.x >> 5] = q; }
    __syncthreads();
    if (threadIdx.x == 0) {
        float S = 0.f, Q = 0.f;
#pragma unroll
        for (int i = 0; i < 8; ++i) { S += rs[i]; Q += rq[i]; }
        float mu  = S * (1.f / CC);
        float var = Q * (1.f / CC) - mu * mu;
        smu = mu;
        sivar = rsqrtf(var + 1e-6f);
    }
    __syncthreads();

    float t = (s - smu) * sivar * lng[c] + lnb[c];
    float g = 0.5f * t * (1.f + erff(t * 0.70710678118654752f));
    size_t o = (size_t)pix * CC + c;
    x1[o]  = g;
    x1h[o] = (_Float16)g;
}

// ---------------------------------------------------------------------------
// Softmax over P=9 per (pixel, group), in place on (M x 72) logits.
// ---------------------------------------------------------------------------
__global__ void softmax9(float* __restrict__ mv) {
    int idx = blockIdx.x * blockDim.x + threadIdx.x;   // pix*G + g
    if (idx >= MROWS * GG) return;
    int g   = idx & (GG - 1);
    int pix = idx >> 3;
    float* p = mv + (size_t)pix * (GG * PP) + g * PP;
    float mx = p[0];
#pragma unroll
    for (int i = 1; i < PP; ++i) mx = fmaxf(mx, p[i]);
    float e[PP], s = 0.f;
#pragma unroll
    for (int i = 0; i < PP; ++i) { e[i] = __expf(p[i] - mx); s += e[i]; }
    float inv = 1.f / s;
#pragma unroll
    for (int i = 0; i < PP; ++i) p[i] = e[i] * inv;
}

// ---------------------------------------------------------------------------
// DCNv3 core. One wave32 per (pixel, group); lane = channel within group
// (Cg == 32 == wave size). Padded-grid sampling collapses analytically to
// unpadded coords: px = w + 2*kx + 2*off_x (pad ring contributes zero).
// ---------------------------------------------------------------------------
__device__ __forceinline__ float samp(const float* __restrict__ xv,
                                      int n, int iy, int ix, int c) {
    if ((unsigned)ix >= (unsigned)WW || (unsigned)iy >= (unsigned)HH) return 0.f;
    return xv[((size_t)((n * HH + iy) * WW + ix)) * CC + c];
}

__global__ __launch_bounds__(256)
void dcnv3_core(const float* __restrict__ xv, const float* __restrict__ off,
                const float* __restrict__ mk,
                float* __restrict__ y, _Float16* __restrict__ yh) {
    const int wid  = blockIdx.x * 8 + (threadIdx.x >> 5);
    const int lane = threadIdx.x & 31;
    const int g    = wid & (GG - 1);
    const int pix  = wid >> 3;
    const int w    = pix & 63;
    const int h    = (pix >> 6) & 63;
    const int n    = pix >> 12;
    const int c    = g * 32 + lane;

    const float* ob = off + ((size_t)pix * GG + g) * (PP * 2);
    const float* mb = mk  + (size_t)pix * (GG * PP) + g * PP;

    float acc = 0.f;
#pragma unroll
    for (int p = 0; p < PP; ++p) {
        const int   kx = p / 3 - 1;           // torch meshgrid ij: x outer
        const int   ky = p % 3 - 1;           // y inner
        const float ox = ob[p * 2 + 0];
        const float oy = ob[p * 2 + 1];
        const float fx = (float)w + 2.f * (float)kx + 2.f * ox;
        const float fy = (float)h + 2.f * (float)ky + 2.f * oy;
        const float x0 = floorf(fx), y0 = floorf(fy);
        const float wx = fx - x0,    wy = fy - y0;
        const int   ix = (int)x0,    iy = (int)y0;
        const float m  = mb[p];

        const float v00 = samp(xv, n, iy,     ix,     c);
        const float v01 = samp(xv, n, iy,     ix + 1, c);
        const float v10 = samp(xv, n, iy + 1, ix,     c);
        const float v11 = samp(xv, n, iy + 1, ix + 1, c);
        const float val = (v00 * (1.f - wx) + v01 * wx) * (1.f - wy)
                        + (v10 * (1.f - wx) + v11 * wx) * wy;
        acc += m * val;
    }
    size_t o = (size_t)pix * CC + c;
    y[o]  = acc;
    yh[o] = (_Float16)acc;
}

// ---------------------------------------------------------------------------
// Host-side launch
// ---------------------------------------------------------------------------
extern "C" void kernel_launch(void* const* d_in, const int* in_sizes, int n_in,
                              void* d_out, int out_size, void* d_ws, size_t ws_size,
                              hipStream_t stream) {
    (void)in_sizes; (void)n_in; (void)out_size; (void)ws_size;
    const float* inputs = (const float*)d_in[0];
    const float* w_in   = (const float*)d_in[1];
    const float* b_in   = (const float*)d_in[2];
    const float* dw_k   = (const float*)d_in[3];
    const float* dw_b   = (const float*)d_in[4];
    const float* ln_g   = (const float*)d_in[5];
    const float* ln_b   = (const float*)d_in[6];
    const float* w_off  = (const float*)d_in[7];
    const float* b_off  = (const float*)d_in[8];
    const float* w_mask = (const float*)d_in[9];
    const float* b_mask = (const float*)d_in[10];
    const float* w_out  = (const float*)d_in[11];
    const float* b_out  = (const float*)d_in[12];
    float* out = (float*)d_out;

    const size_t SZ = (size_t)MROWS * CC;     // 4,194,304 elems
    char* ws = (char*)d_ws;
    size_t o = 0;
    auto alloc = [&](size_t bytes) { void* p = ws + o; o += (bytes + 255) & ~(size_t)255; return p; };

    float*    x_f32  = (float*)   alloc(SZ * 4);            // NHWC input (reused as y)
    _Float16* x_h    = (_Float16*)alloc(SZ * 2);            // f16 input (reused as y_h)
    float*    xv     = (float*)   alloc(SZ * 4);            // value features
    float*    x1     = (float*)   alloc(SZ * 4);            // conv+LN+GELU
    _Float16* x1_h   = (_Float16*)alloc(SZ * 2);
    float*    offv   = (float*)   alloc((size_t)MROWS * 144 * 4);
    float*    maskv  = (float*)   alloc((size_t)MROWS * 72 * 4);
    _Float16* wt_in  = (_Float16*)alloc((size_t)256 * CC * 2);
    _Float16* wt_off = (_Float16*)alloc((size_t)160 * CC * 2);  // 144 -> pad 160
    _Float16* wt_msk = (_Float16*)alloc((size_t)96  * CC * 2);  // 72  -> pad 96
    _Float16* wt_out = (_Float16*)alloc((size_t)256 * CC * 2);
    float*    yv     = x_f32;                                // reuse
    _Float16* y_h    = x_h;                                  // reuse

    // 1) weight transpose/convert (Bt: Npad x K, f16; Npad multiple of 32)
    prep_w<<<(256 * CC + 255) / 256, 256, 0, stream>>>(w_in,   wt_in,  256, 256, CC);
    prep_w<<<(160 * CC + 255) / 256, 256, 0, stream>>>(w_off,  wt_off, 144, 160, CC);
    prep_w<<<(96  * CC + 255) / 256, 256, 0, stream>>>(w_mask, wt_msk,  72,  96, CC);
    prep_w<<<(256 * CC + 255) / 256, 256, 0, stream>>>(w_out,  wt_out, 256, 256, CC);

    // 2) NCHW -> NHWC
    nchw_to_nhwc<<<(int)(SZ / 256), 256, 0, stream>>>(inputs, x_f32, x_h);

    const int Mt2 = MROWS / 32;   // 512 macro-row tiles

    // 3) xv = x @ w_in + b_in        (Npad = 256 -> 8 macro tiles)
    gemm_wmma_f16<<<Mt2 * 8 / 8, 256, 0, stream>>>(x_h, wt_in, b_in, xv, 256, 8, CC, 0);

    // 4) depthwise conv + LN + GELU
    conv_ln_gelu<<<MROWS, 256, 0, stream>>>(x_f32, dw_k, dw_b, ln_g, ln_b, x1, x1_h);

    // 5) offset = x1 @ w_off + b_off (Npad = 160 -> 5 macro tiles)
    gemm_wmma_f16<<<Mt2 * 5 / 8, 256, 0, stream>>>(x1_h, wt_off, b_off, offv, 144, 5, CC, 0);

    // 6) mask logits = x1 @ w_mask + b_mask (Npad = 96 -> 3 macro tiles)
    gemm_wmma_f16<<<Mt2 * 3 / 8, 256, 0, stream>>>(x1_h, wt_msk, b_mask, maskv, 72, 3, CC, 0);

    // 7) softmax over P=9
    softmax9<<<(MROWS * GG + 255) / 256, 256, 0, stream>>>(maskv);

    // 8) DCNv3 sampling core: wave per (pixel, group)
    dcnv3_core<<<MROWS * GG / 8, 256, 0, stream>>>(xv, offv, maskv, yv, y_h);

    // 9) out = y @ w_out + b_out, scattered to NCHW
    gemm_wmma_f16<<<Mt2 * 8 / 8, 256, 0, stream>>>(y_h, wt_out, b_out, out, 256, 8, CC, 1);
}
